// DCNv2_Circle8_85117661872715
// MI455X (gfx1250) — matrix operand order repaired and checked by reading, not compiled, not run on hardware
//
#include <hip/hip_runtime.h>
#include <math.h>

typedef __attribute__((ext_vector_type(16))) _Float16 v16h;
typedef __attribute__((ext_vector_type(2)))  _Float16 v2h;
typedef __attribute__((ext_vector_type(8)))  float    v8f;

#define HWSZ   16384          // H*W = 128*128
#define WIMG   128
#define CIN    64
#define COUT   64
#define KDIM   576            // CIN * 9
#define KSTEPS 18             // KDIM / 32
#define MT     32             // pixels per block (2 M-tiles of 16)
#define SOSTR  68             // padded out-accum row stride in floats

__constant__ int c_perm[8][9] = {
    {0,1,2,3,4,5,6,7,8},
    {3,0,1,6,4,2,7,8,5},
    {6,3,0,7,4,1,8,5,2},
    {7,6,3,8,4,0,5,2,1},
    {8,7,6,5,4,3,2,1,0},
    {5,8,7,2,4,6,1,0,3},
    {2,5,8,1,4,7,0,3,6},
    {1,2,5,0,4,8,3,6,7}
};

// ---------------- Kernel 1a: per-(b,c) plane sum (deterministic) -------------
__global__ __launch_bounds__(256) void k_pool_partial(const float* __restrict__ input,
                                                      float* __restrict__ P) {
    __shared__ float red[256];
    const int blk = blockIdx.x;                 // b*64 + c
    const float* src = input + (size_t)blk * HWSZ;
    float s = 0.f;
    for (int i = threadIdx.x; i < HWSZ; i += 256) s += src[i];
    red[threadIdx.x] = s;
    __syncthreads();
    for (int off = 128; off > 0; off >>= 1) {
        if (threadIdx.x < off) red[threadIdx.x] += red[threadIdx.x + off];
        __syncthreads();
    }
    if (threadIdx.x == 0) P[blk] = red[0];
}

// ---------------- Kernel 1b: finish mean, FC(4x64) + sigmoid -> r[4] ---------
__global__ __launch_bounds__(64) void k_fc(const float* __restrict__ P,
                                           const float* __restrict__ fc_w,
                                           const float* __restrict__ fc_b,
                                           float* __restrict__ r) {
    __shared__ float s[64];
    const int c = threadIdx.x;
    s[c] = (P[c] + P[64 + c]) * (1.0f / (float)HWSZ);   // sum over b of mean_hw
    __syncthreads();
    if (c < 4) {
        float acc = fc_b[c];
        #pragma unroll 16
        for (int j = 0; j < 64; ++j) acc += s[j] * fc_w[c * 64 + j];
        r[c] = 1.0f / (1.0f + expf(-acc));
    }
}

// ---------------- Kernel 2: circle-blend weights -> f16 B-fragment layout ----
__global__ __launch_bounds__(256) void k_blend(const float* __restrict__ weight,
                                               const float* __restrict__ r,
                                               _Float16* __restrict__ wqB) {
    const int t = blockIdx.x * 256 + threadIdx.x;     // = o*64 + c, 0..4095
    if (t >= 4096) return;
    const int o = t >> 6;
    const int c = t & 63;
    float wr[9];
    const float* wp = weight + (size_t)t * 9;
    #pragma unroll
    for (int j = 0; j < 9; ++j) wr[j] = wp[j];

    const float S  = 0.70710678118654752440f;
    const float T  = 1.0f - S;
    float f0 = S * wr[0] + T * wr[1];
    float f1 = S * wr[3] + T * wr[4];
    float w0 = S * f0 + T * f1;
    float f2 = T * wr[1] + S * wr[2];
    float f3 = T * wr[4] + S * wr[5];
    float w2 = S * f2 + T * f3;
    float f5 = S * wr[6] + T * wr[7];
    float w6 = T * f1 + S * f5;          // f4 == f1
    float f7 = T * wr[7] + S * wr[8];
    float w8 = T * f3 + S * f7;          // f6 == f3
    float base[9] = {w0, wr[1], w2, wr[3], wr[4], wr[5], w6, wr[7], w8};

    for (int g = 0; g < 8; ++g) {
        const float f = (g & 1) ? 1.0f : r[g >> 1];
        const int oglob = g * 64 + o;
        const int ntile = oglob >> 4;
        const int n     = oglob & 15;
        #pragma unroll
        for (int k = 0; k < 9; ++k) {
            const int p   = c_perm[g][k];
            const float v = base[p] * f + wr[p] * (1.0f - f);
            const int K     = c * 9 + k;
            const int kstep = K >> 5;
            const int kk    = K & 31;
            const int lane  = n + (((kk >> 3) & 1) << 4);
            const int half  = (kk & 7) | ((kk >> 4) << 3);
            wqB[(((size_t)(ntile * KSTEPS + kstep) * 32) + lane) * 16 + half] = (_Float16)v;
        }
    }
}

// ---------------- Kernel 3: deformable sample + WMMA GEMM + group combine ----
// LDS A tile is stored directly in WMMA A-fragment order:
//   chunk(ks, mtile, grp, Mrow) of 16 halves; lane reads one 32B chunk per frag.
//   kk -> (grp, e):  grp = (kk>>3)&1,  e = (kk&7) | ((kk>>4)<<3)
__global__ __launch_bounds__(256) void k_dcn(const float* __restrict__ input,
                                             const float* __restrict__ offset,
                                             const float* __restrict__ maskp,
                                             const float* __restrict__ wc8,
                                             const float* __restrict__ bias,
                                             const _Float16* __restrict__ wqB,
                                             float* __restrict__ out) {
    __shared__ __align__(32) _Float16 sA[KSTEPS * 2 * 2 * 16 * 16]; // 36864 B
    __shared__ float sWgt[MT * 9 * 4];                              //  4608 B
    __shared__ int   sIdx[MT * 9 * 4];                              //  4608 B
    __shared__ float sW8[MT * 8];                                   //  1024 B
    __shared__ float sOut[MT * SOSTR];                              //  8704 B

    const int tid     = threadIdx.x;
    const int pixbase = blockIdx.x * MT;                  // 1024 blocks
    const int b       = pixbase >> 14;                    // / HWSZ
    const int hwbase  = pixbase & (HWSZ - 1);             // tile stays in one row

    // ---- Phase 1: bilinear coefficients (mask folded in), w_c8, zero accum --
    for (int t2 = tid; t2 < MT * 9; t2 += 256) {
        const int p  = t2 / 9;
        const int k  = t2 % 9;
        const int hw = hwbase + p;
        const int h  = hw >> 7;
        const int w  = hw & 127;
        const float oy = offset[((size_t)(b * 18 + 2 * k))     * HWSZ + hw];
        const float ox = offset[((size_t)(b * 18 + 2 * k + 1)) * HWSZ + hw];
        const float mk = maskp [((size_t)(b * 9  + k))         * HWSZ + hw];
        const float py = (float)h + (float)(k / 3) - 1.0f + oy;
        const float px = (float)w + (float)(k % 3) - 1.0f + ox;
        const float y0f = floorf(py), x0f = floorf(px);
        const float ly = py - y0f, lx = px - x0f;
        const int y0 = (int)y0f, x0 = (int)x0f;
        #pragma unroll
        for (int i = 0; i < 4; ++i) {
            const int dy = i >> 1, dx = i & 1;
            const int yi = y0 + dy, xi = x0 + dx;
            const bool valid = (yi >= 0) & (yi < WIMG) & (xi >= 0) & (xi < WIMG);
            float wgt = (dy ? ly : 1.0f - ly) * (dx ? lx : 1.0f - lx);
            wgt = valid ? wgt * mk : 0.0f;
            int yc = yi < 0 ? 0 : (yi > WIMG - 1 ? WIMG - 1 : yi);
            int xc = xi < 0 ? 0 : (xi > WIMG - 1 ? WIMG - 1 : xi);
            sWgt[t2 * 4 + i] = wgt;
            sIdx[t2 * 4 + i] = yc * WIMG + xc;
        }
    }
    for (int t2 = tid; t2 < MT * 8; t2 += 256) {
        const int p = t2 >> 3, g = t2 & 7;
        sW8[t2] = wc8[((size_t)(b * 8 + g)) * HWSZ + hwbase + p];
    }
    for (int t2 = tid; t2 < MT * SOSTR; t2 += 256) sOut[t2] = 0.0f;
    __syncthreads();

    // ---- Phase 2: gather; two consecutive K per thread, one b32 LDS store ---
    const float* inb = input + (size_t)b * CIN * HWSZ;
    for (int j = tid; j < MT * (KDIM / 2); j += 256) {    // 36 iterations
        const int p  = j / (KDIM / 2);                    // pixel 0..31
        const int K  = (j % (KDIM / 2)) * 2;              // even K
        float v[2];
        #pragma unroll
        for (int u = 0; u < 2; ++u) {
            const int Ku = K + u;
            const int c  = Ku / 9;
            const int k  = Ku - 9 * c;
            const int pk = p * 9 + k;
            const float* ip = inb + (size_t)c * HWSZ;
            const int*   xi = &sIdx[pk * 4];
            const float* xw = &sWgt[pk * 4];
            v[u] = xw[0] * ip[xi[0]] + xw[1] * ip[xi[1]]
                 + xw[2] * ip[xi[2]] + xw[3] * ip[xi[3]];
        }
        // fragment-layout address (e is even, so the pair is contiguous)
        const int ks  = K >> 5;
        const int kk  = K & 31;
        const int grp = (kk >> 3) & 1;
        const int e   = (kk & 7) | ((kk >> 4) << 3);
        const int mtile = p >> 4, Mrow = p & 15;
        const int idx = ((((ks * 2 + mtile) * 2 + grp) << 4) + Mrow) * 16 + e;
        v2h pair; pair[0] = (_Float16)v[0]; pair[1] = (_Float16)v[1];
        *(v2h*)&sA[idx] = pair;
    }
    __syncthreads();

    // ---- Phase 3: WMMA GEMM. Wave `wid` owns output group g = wid -----------
    const int lane = tid & 31;
    const int wid  = tid >> 5;
    const int grp  = lane >> 4;
    const int Mrow = lane & 15;

    v8f acc[2][4];
    #pragma unroll
    for (int m = 0; m < 2; ++m)
        #pragma unroll
        for (int nt = 0; nt < 4; ++nt)
            #pragma unroll
            for (int e = 0; e < 8; ++e) acc[m][nt][e] = 0.0f;

    for (int ks = 0; ks < KSTEPS; ++ks) {
        const v16h a0 = *(const v16h*)&sA[((((ks * 2 + 0) * 2 + grp) << 4) + Mrow) << 4];
        const v16h a1 = *(const v16h*)&sA[((((ks * 2 + 1) * 2 + grp) << 4) + Mrow) << 4];
        if (ks + 1 < KSTEPS)
            __builtin_prefetch(wqB + (((size_t)(wid * 4) * KSTEPS + ks + 1) * 32 + lane) * 16, 0, 3);
        #pragma unroll
        for (int nt = 0; nt < 4; ++nt) {
            const int ntile = wid * 4 + nt;
            const v16h bf = *(const v16h*)(wqB + (((size_t)ntile * KSTEPS + ks) * 32 + lane) * 16);
            acc[0][nt] = __builtin_amdgcn_wmma_f32_16x16x32_f16(
                false, a0, false, bf, (short)0, acc[0][nt], false, false);
            acc[1][nt] = __builtin_amdgcn_wmma_f32_16x16x32_f16(
                false, a1, false, bf, (short)0, acc[1][nt], false, false);
        }
    }

    // ---- Phase 4: bias + w_c8 scaling, deterministic staged cross-wave sum --
    const int msel = grp << 3;                            // 0 or 8 row offset
    for (int gp = 0; gp < 8; ++gp) {
        if (wid == gp) {
            #pragma unroll
            for (int m = 0; m < 2; ++m) {
                #pragma unroll
                for (int nt = 0; nt < 4; ++nt) {
                    const int co = nt * 16 + Mrow;
                    const float bs = bias[wid * 64 + co];
                    #pragma unroll
                    for (int rr = 0; rr < 8; ++rr) {
                        const int M = m * 16 + rr + msel;
                        sOut[M * SOSTR + co] += (acc[m][nt][rr] + bs) * sW8[M * 8 + wid];
                    }
                }
            }
        }
        __syncthreads();
    }

    // ---- coalesced store: out[b, co, hwbase + p] ----------------------------
    for (int j = tid; j < MT * COUT; j += 256) {
        const int p  = j & 31;
        const int co = j >> 5;
        out[((size_t)(b * COUT + co)) * HWSZ + hwbase + p] = sOut[p * SOSTR + co];
    }
}

// ---------------------------------------------------------------------------
extern "C" void kernel_launch(void* const* d_in, const int* in_sizes, int n_in,
                              void* d_out, int out_size, void* d_ws, size_t ws_size,
                              hipStream_t stream) {
    const float* input  = (const float*)d_in[0];   // (2,64,128,128)
    const float* offset = (const float*)d_in[1];   // (2,18,128,128)
    const float* maskp  = (const float*)d_in[2];   // (2,9,128,128)
    const float* wc8    = (const float*)d_in[3];   // (2,8,128,128)
    const float* weight = (const float*)d_in[4];   // (64,64,3,3)
    const float* bias   = (const float*)d_in[5];   // (512,)
    const float* fc_w   = (const float*)d_in[6];   // (4,64)
    const float* fc_b   = (const float*)d_in[7];   // (4,)
    float* out = (float*)d_out;                    // (2,64,128,128)

    char* ws = (char*)d_ws;
    float*    P   = (float*)ws;                    // 128 floats: per-(b,c) plane sums
    float*    r   = (float*)(ws + 512);            // 4 floats: sigmoid gates
    _Float16* wqB = (_Float16*)(ws + 1024);        // 512x576 f16 in B-fragment layout (576 KB)

    k_pool_partial<<<128, 256, 0, stream>>>(input, P);
    k_fc<<<1, 64, 0, stream>>>(P, fc_w, fc_b, r);
    k_blend<<<16, 256, 0, stream>>>(weight, r, wqB);
    k_dcn<<<1024, 256, 0, stream>>>(input, offset, maskp, wc8, bias, wqB, out);
}
